// CustomFullyConnectedLayer_11690900980059
// MI455X (gfx1250) — matrix operand order repaired and checked by reading, not compiled, not run on hardware
//
#include <hip/hip_runtime.h>

typedef __attribute__((ext_vector_type(16))) __bf16 v16bf;
typedef __attribute__((ext_vector_type(8)))  float  v8f;

#define IN_F   4096
#define OUT_F  4096
#define BATCH  8192
#define NMOD   4095u

#define BM 128
#define BN 128
#define BK 32
#define XST 36   // padded f32 row stride for X tile in fallback (144B, 16B aligned)
#define BST 40   // padded bf16 row stride (80B, 16B aligned, conflict-free)
#define KTILES (IN_F / BK)

static __device__ __forceinline__ unsigned short f32_to_bf16(float f) {
  unsigned int u = __float_as_uint(f);
  u += 0x7FFFu + ((u >> 16) & 1u);           // round-to-nearest-even
  return (unsigned short)(u >> 16);
}
static __device__ __forceinline__ float bf16_to_f32(unsigned short h) {
  return __uint_as_float(((unsigned int)h) << 16);
}
// hi = truncate-to-bf16(f) (1 shift), lo = RNE-bf16(f - hi); total err ~2^-17 rel
static __device__ __forceinline__ void split_bf16(float f, unsigned short& h,
                                                  unsigned short& l) {
  const unsigned int u = __float_as_uint(f);
  h = (unsigned short)(u >> 16);
  l = f32_to_bf16(f - __uint_as_float(u & 0xFFFF0000u));
}
static __device__ __forceinline__ unsigned lds_off(const void* p) {
  return (unsigned)(reinterpret_cast<size_t>(p));  // low 32b of flat LDS ptr
}
// gfx1250 async copy: 16B per lane, global -> LDS, tracked by ASYNCcnt.
static __device__ __forceinline__ void async_ld_b128(unsigned lds_byte, const void* gp) {
  asm volatile("global_load_async_to_lds_b128 %0, %1, off"
               :: "v"(lds_byte), "v"((unsigned long long)reinterpret_cast<size_t>(gp))
               : "memory");
}
static __device__ __forceinline__ void wait_async0() {
#if __has_builtin(__builtin_amdgcn_s_wait_asynccnt)
  __builtin_amdgcn_s_wait_asynccnt(0);
#else
  asm volatile("s_wait_asynccnt 0x0" ::: "memory");
#endif
}

// ---------------------------------------------------------------------------
// Kernel 1: Dykstra soft-top-k on alpha (n = 4096, 50 iterations).
// ---------------------------------------------------------------------------
__global__ __launch_bounds__(1024) void dykstra_kernel(
    const float* __restrict__ alpha, const int* __restrict__ Kp,
    float* __restrict__ at) {
  __shared__ float sdata[1024];
  const int tid = threadIdx.x;
  const float kf = (float)Kp[0];
  float z[4], q[4];
#pragma unroll
  for (int j = 0; j < 4; ++j) {
    z[j] = alpha[tid * 4 + j] * 100.0f;  // x / l, l = 0.01
    q[j] = 0.0f;
  }
  float p = 0.0f;
  for (int it = 0; it < 50; ++it) {
    float part = 0.0f;
#pragma unroll
    for (int j = 0; j < 4; ++j) part += z[j] + p;
    sdata[tid] = part;
    __syncthreads();
    for (int s = 512; s > 0; s >>= 1) {
      if (tid < s) sdata[tid] += sdata[tid + s];
      __syncthreads();
    }
    const float S = sdata[0];
    __syncthreads();
    const float corr = (kf - S) * (1.0f / 4096.0f);
#pragma unroll
    for (int j = 0; j < 4; ++j) {
      float z1 = z[j] + p + corr;
      float w2 = z1 + q[j];
      float z2 = fminf(fmaxf(w2, 0.0f), 1.0f);
      q[j] = w2 - z2;
      z[j] = z2;
    }
    p = -corr;
  }
#pragma unroll
  for (int j = 0; j < 4; ++j) at[tid * 4 + j] = z[j];
}

// ---------------------------------------------------------------------------
// Kernel 2: W[r,c] = at[(r-c)&4095] * V[(r-c)&4095, c] -> bf16 hi/lo planes.
// ---------------------------------------------------------------------------
__global__ __launch_bounds__(256) void build_w_kernel(
    const float* __restrict__ V, const float* __restrict__ at,
    unsigned short* __restrict__ Wh, unsigned short* __restrict__ Wl) {
  __shared__ float sV[64 * 32];
  const int tid = threadIdx.x;
  const int r0 = blockIdx.x * 32;
  const int c0 = blockIdx.y * 32;
  const unsigned ddmin = ((unsigned)(r0 - c0 - 31)) & NMOD;

  for (int s = tid; s < 64 * 32; s += 256) {
    const int dd = s >> 5;
    const int j  = s & 31;
    const unsigned d = (ddmin + (unsigned)dd) & NMOD;
    sV[s] = at[d] * V[(size_t)d * IN_F + (unsigned)(c0 + j)];
  }
  __syncthreads();

#pragma unroll
  for (int pass = 0; pass < 4; ++pass) {
    const int rr = pass * 8 + (tid >> 5);
    const int cc = tid & 31;
    const int dd = rr - cc + 31;               // 0..62
    const float w = sV[dd * 32 + cc];          // lane stride -31: conflict-free
    unsigned short hi, lo;
    split_bf16(w, hi, lo);
    const size_t o = (size_t)(r0 + rr) * IN_F + (size_t)(c0 + cc);
    Wh[o] = hi;
    Wl[o] = lo;
  }
}

// ---------------------------------------------------------------------------
// Kernel 2b: pre-split x into bf16 hi/lo planes (one pass; removes all
// conversion VALU from the GEMM hot loop).
// ---------------------------------------------------------------------------
__global__ __launch_bounds__(256) void split_x_kernel(
    const float* __restrict__ xg,
    unsigned short* __restrict__ Xh, unsigned short* __restrict__ Xl) {
  const size_t base = ((size_t)blockIdx.x * 256 + threadIdx.x) * 8;
#pragma unroll
  for (int p = 0; p < 2; ++p) {
    const float4 v = *(const float4*)(xg + base + p * 4);
    unsigned short h0, h1, h2, h3, l0, l1, l2, l3;
    split_bf16(v.x, h0, l0);
    split_bf16(v.y, h1, l1);
    split_bf16(v.z, h2, l2);
    split_bf16(v.w, h3, l3);
    uint2 hv, lv;
    hv.x = (unsigned)h0 | ((unsigned)h1 << 16);
    hv.y = (unsigned)h2 | ((unsigned)h3 << 16);
    lv.x = (unsigned)l0 | ((unsigned)l1 << 16);
    lv.y = (unsigned)l2 | ((unsigned)l3 << 16);
    *(uint2*)(Xh + base + p * 4) = hv;
    *(uint2*)(Xl + base + p * 4) = lv;
  }
}

// ---------------------------------------------------------------------------
// Kernel 3 (main): out = x @ W^T with all four operands as bf16 planes.
// Block: 256 threads (8 wave32), 128x128 tile; wave 32(M) x 64(N).
// Double-buffered LDS filled by GLOBAL_LOAD_ASYNC_TO_LDS_B128; hot loop is
// ds_load_b128 fragment loads + 24 v_wmma per wave per K-step, ~no VALU.
// ---------------------------------------------------------------------------
__global__ __launch_bounds__(256) void gemm_planes_kernel(
    const unsigned short* __restrict__ Xh, const unsigned short* __restrict__ Xl,
    const unsigned short* __restrict__ Wh, const unsigned short* __restrict__ Wl,
    float* __restrict__ out) {
  __shared__ __align__(16) unsigned short sAh[2][BM * BST];
  __shared__ __align__(16) unsigned short sAl[2][BM * BST];
  __shared__ __align__(16) unsigned short sBh[2][BN * BST];
  __shared__ __align__(16) unsigned short sBl[2][BN * BST];

  const int tid  = threadIdx.x;
  const int lane = tid & 31;
  const int wid  = tid >> 5;
  const int wm   = wid & 3;
  const int wn   = wid >> 2;

  const int m0 = blockIdx.x * BM;
  const int n0 = blockIdx.y * BN;

  const int wrow = tid >> 2, wjg = tid & 3;   // 4 b128-granules per 32-elem row

  v8f acc[2][4];
  const v8f vzero = {0.f, 0.f, 0.f, 0.f, 0.f, 0.f, 0.f, 0.f};
#pragma unroll
  for (int t = 0; t < 2; ++t)
#pragma unroll
    for (int j = 0; j < 4; ++j) acc[t][j] = vzero;

  union BFrag { uint4 q[2]; v16bf v; };

  auto stage = [&](int k0, int buf) {
#pragma unroll
    for (int p = 0; p < 2; ++p) {                       // 512 granules / plane
      const int row = wrow + p * 64;
      const unsigned soff = (unsigned)(row * BST + wjg * 8);
      const size_t ga = (size_t)(m0 + row) * IN_F + k0 + wjg * 8;
      const size_t gb = (size_t)(n0 + row) * IN_F + k0 + wjg * 8;
      async_ld_b128(lds_off(&sAh[buf][soff]), Xh + ga);
      async_ld_b128(lds_off(&sAl[buf][soff]), Xl + ga);
      async_ld_b128(lds_off(&sBh[buf][soff]), Wh + gb);
      async_ld_b128(lds_off(&sBl[buf][soff]), Wl + gb);
    }
  };

  stage(0, 0);

  for (int kt = 0; kt < KTILES; ++kt) {
    wait_async0();
    __syncthreads();
    if (kt + 1 < KTILES) stage((kt + 1) * BK, (kt + 1) & 1);
    const int buf = kt & 1;

    // A frags (ISA 7.12.2): row = lane&15; lane<16: K{0..7,16..23} else K{8..15,24..31}
    const int abase = (lane & 15);
    const int akoff = (lane >> 4) * 8;
    BFrag ah[2], al[2];
#pragma unroll
    for (int t = 0; t < 2; ++t) {
      const unsigned short* ph = &sAh[buf][(wm * 32 + t * 16 + abase) * BST + akoff];
      ah[t].q[0] = *(const uint4*)ph;
      ah[t].q[1] = *(const uint4*)(ph + 16);
      const unsigned short* pl = &sAl[buf][(wm * 32 + t * 16 + abase) * BST + akoff];
      al[t].q[0] = *(const uint4*)pl;
      al[t].q[1] = *(const uint4*)(pl + 16);
    }
    // B frags: col = lane&15; lane<16: K{0..7,16..23} else K{8..15,24..31}
    const int bkoff = (lane >> 4) * 8;
    BFrag bh[4], bl[4];
#pragma unroll
    for (int j = 0; j < 4; ++j) {
      const unsigned short* pb = &sBh[buf][(wn * 64 + j * 16 + (lane & 15)) * BST + bkoff];
      bh[j].q[0] = *(const uint4*)pb;
      bh[j].q[1] = *(const uint4*)(pb + 16);
      const unsigned short* pq = &sBl[buf][(wn * 64 + j * 16 + (lane & 15)) * BST + bkoff];
      bl[j].q[0] = *(const uint4*)pq;
      bl[j].q[1] = *(const uint4*)(pq + 16);
    }

#pragma unroll
    for (int t = 0; t < 2; ++t)
#pragma unroll
      for (int j = 0; j < 4; ++j) {
        acc[t][j] = __builtin_amdgcn_wmma_f32_16x16x32_bf16(
            false, ah[t].v, false, bh[j].v, (short)0, acc[t][j], false, false);
        acc[t][j] = __builtin_amdgcn_wmma_f32_16x16x32_bf16(
            false, ah[t].v, false, bl[j].v, (short)0, acc[t][j], false, false);
        acc[t][j] = __builtin_amdgcn_wmma_f32_16x16x32_bf16(
            false, al[t].v, false, bh[j].v, (short)0, acc[t][j], false, false);
      }
  }

  const int mrow = m0 + wm * 32;
  const int ncol = n0 + wn * 64;
#pragma unroll
  for (int t = 0; t < 2; ++t)
#pragma unroll
    for (int j = 0; j < 4; ++j) {
      const int rbase = mrow + t * 16 + 8 * (lane >> 4);
      const int cbase = ncol + j * 16 + (lane & 15);
#pragma unroll
      for (int v = 0; v < 8; ++v)
        out[(size_t)(rbase + v) * OUT_F + cbase] = acc[t][j][v];
    }
}

// ---------------------------------------------------------------------------
// Kernel 3 (fallback, small workspace): X staged as f32, split at frag-load.
// ---------------------------------------------------------------------------
__global__ __launch_bounds__(256) void gemm_conv_kernel(
    const float* __restrict__ xg,
    const unsigned short* __restrict__ Wh,
    const unsigned short* __restrict__ Wl,
    float* __restrict__ out) {
  __shared__ __align__(16) float          sX[2][BM * XST];
  __shared__ __align__(16) unsigned short sBh[2][BN * BST];
  __shared__ __align__(16) unsigned short sBl[2][BN * BST];

  const int tid  = threadIdx.x;
  const int lane = tid & 31;
  const int wid  = tid >> 5;
  const int wm   = wid & 3;
  const int wn   = wid >> 2;
  const int m0 = blockIdx.x * BM;
  const int n0 = blockIdx.y * BN;
  const int xrow = tid >> 3, xkg = tid & 7;
  const int wrow = tid >> 2, wjg = tid & 3;

  v8f acc[2][4];
  const v8f vzero = {0.f, 0.f, 0.f, 0.f, 0.f, 0.f, 0.f, 0.f};
#pragma unroll
  for (int t = 0; t < 2; ++t)
#pragma unroll
    for (int j = 0; j < 4; ++j) acc[t][j] = vzero;

  union BFrag { uint4 q[2]; v16bf v; unsigned short s[16]; };
  union F32x8 { uint4 q[2]; float f[8]; };

  auto stage = [&](int k0, int buf) {
#pragma unroll
    for (int p = 0; p < 4; ++p) {
      const int row = xrow + p * 32;
      async_ld_b128(lds_off(&sX[buf][row * XST + xkg * 4]),
                    xg + (size_t)(m0 + row) * IN_F + k0 + xkg * 4);
    }
#pragma unroll
    for (int p = 0; p < 2; ++p) {
      const int row = wrow + p * 64;
      async_ld_b128(lds_off(&sBh[buf][row * BST + wjg * 8]),
                    Wh + (size_t)(n0 + row) * IN_F + k0 + wjg * 8);
      async_ld_b128(lds_off(&sBl[buf][row * BST + wjg * 8]),
                    Wl + (size_t)(n0 + row) * IN_F + k0 + wjg * 8);
    }
  };

  stage(0, 0);

  for (int kt = 0; kt < KTILES; ++kt) {
    wait_async0();
    __syncthreads();
    if (kt + 1 < KTILES) stage((kt + 1) * BK, (kt + 1) & 1);
    const int buf = kt & 1;

    const int arow  = lane & 15;
    const int akoff = (lane >> 4) * 8;
    BFrag ah[2], al[2];
#pragma unroll
    for (int t = 0; t < 2; ++t) {
      const float* px = &sX[buf][(wm * 32 + t * 16 + arow) * XST + akoff];
      F32x8 c0, c1;
      c0.q[0] = *(const uint4*)(px);
      c0.q[1] = *(const uint4*)(px + 4);
      c1.q[0] = *(const uint4*)(px + 16);
      c1.q[1] = *(const uint4*)(px + 20);
#pragma unroll
      for (int e = 0; e < 8; ++e) {
        unsigned short h, l;
        split_bf16(c0.f[e], h, l);
        ah[t].s[e] = h;  al[t].s[e] = l;
        split_bf16(c1.f[e], h, l);
        ah[t].s[8 + e] = h;  al[t].s[8 + e] = l;
      }
    }
    const int bkoff = (lane >> 4) * 8;
    BFrag bh[4], bl[4];
#pragma unroll
    for (int j = 0; j < 4; ++j) {
      const unsigned short* pb = &sBh[buf][(wn * 64 + j * 16 + (lane & 15)) * BST + bkoff];
      bh[j].q[0] = *(const uint4*)pb;
      bh[j].q[1] = *(const uint4*)(pb + 16);
      const unsigned short* pq = &sBl[buf][(wn * 64 + j * 16 + (lane & 15)) * BST + bkoff];
      bl[j].q[0] = *(const uint4*)pq;
      bl[j].q[1] = *(const uint4*)(pq + 16);
    }
#pragma unroll
    for (int t = 0; t < 2; ++t)
#pragma unroll
      for (int j = 0; j < 4; ++j) {
        acc[t][j] = __builtin_amdgcn_wmma_f32_16x16x32_bf16(
            false, ah[t].v, false, bh[j].v, (short)0, acc[t][j], false, false);
        acc[t][j] = __builtin_amdgcn_wmma_f32_16x16x32_bf16(
            false, ah[t].v, false, bl[j].v, (short)0, acc[t][j], false, false);
        acc[t][j] = __builtin_amdgcn_wmma_f32_16x16x32_bf16(
            false, al[t].v, false, bh[j].v, (short)0, acc[t][j], false, false);
      }
  }

  const int mrow = m0 + wm * 32;
  const int ncol = n0 + wn * 64;
#pragma unroll
  for (int t = 0; t < 2; ++t)
#pragma unroll
    for (int j = 0; j < 4; ++j) {
      const int rbase = mrow + t * 16 + 8 * (lane >> 4);
      const int cbase = ncol + j * 16 + (lane & 15);
#pragma unroll
      for (int v = 0; v < 8; ++v)
        out[(size_t)(rbase + v) * OUT_F + cbase] = acc[t][j][v];
    }
}

// ---------------------------------------------------------------------------
extern "C" void kernel_launch(void* const* d_in, const int* in_sizes, int n_in,
                              void* d_out, int out_size, void* d_ws, size_t ws_size,
                              hipStream_t stream) {
  const float* x     = (const float*)d_in[0];
  const float* V     = (const float*)d_in[1];
  const float* alpha = (const float*)d_in[2];
  const int*   K     = (const int*)d_in[3];
  float* out = (float*)d_out;

  // ws layout: at[4096] f32 | Wh | Wl (bf16 4096x4096 each) | Xh | Xl (bf16 8192x4096)
  const size_t W_ELEMS = (size_t)OUT_F * IN_F;
  const size_t X_ELEMS = (size_t)BATCH * IN_F;
  float* at = (float*)d_ws;
  unsigned short* Wh = (unsigned short*)((char*)d_ws + 16384);
  unsigned short* Wl = Wh + W_ELEMS;
  unsigned short* Xh = Wl + W_ELEMS;
  unsigned short* Xl = Xh + X_ELEMS;
  const size_t need_full = 16384 + 2 * W_ELEMS * 2 + 2 * X_ELEMS * 2;

  dykstra_kernel<<<1, 1024, 0, stream>>>(alpha, K, at);
  build_w_kernel<<<dim3(OUT_F / 32, IN_F / 32), 256, 0, stream>>>(V, at, Wh, Wl);

  if (ws_size >= need_full) {
    split_x_kernel<<<(int)(X_ELEMS / (256 * 8)), 256, 0, stream>>>(x, Xh, Xl);
    gemm_planes_kernel<<<dim3(BATCH / BM, OUT_F / BN), 256, 0, stream>>>(Xh, Xl, Wh, Wl, out);
  } else {
    gemm_conv_kernel<<<dim3(BATCH / BM, OUT_F / BN), 256, 0, stream>>>(x, Wh, Wl, out);
  }
}
